// PulseRetrievalLossFunction_85212151153245
// MI455X (gfx1250) — matrix operand order repaired and checked by reading, not compiled, not run on hardware
//
#include <hip/hip_runtime.h>

#define NN 1024
#define BB 32
#define PI2F 6.28318530717958647692f
#define ROWS_PER_BLOCK 8

typedef __attribute__((ext_vector_type(16))) _Float16 v16h;
typedef __attribute__((ext_vector_type(8)))  float    v8f;
typedef __attribute__((ext_vector_type(4)))  unsigned int ui32x4;
typedef __attribute__((ext_vector_type(8)))  int          i32x8;
typedef __attribute__((ext_vector_type(4)))  int          i32x4;

__device__ __forceinline__ float2 cmulf(float2 a, float2 b) {
    return make_float2(a.x * b.x - a.y * b.y, a.x * b.y + a.y * b.x);
}
__device__ __forceinline__ float2 caddf(float2 a, float2 b) { return make_float2(a.x + b.x, a.y + b.y); }
__device__ __forceinline__ float2 csubf(float2 a, float2 b) { return make_float2(a.x - b.x, a.y - b.y); }

// ---------------------------------------------------------------------------
// TDM: 1D tile (1024 x 8B) global -> LDS DMA via hand-built D# (ISA ch.8).
// Gated on builtin availability + clang arity difference (5-arg clang-22 /
// 6-arg clang-23); falls back to plain per-lane loads.
// ---------------------------------------------------------------------------
__device__ __forceinline__ void tdm_load_row_f2(float2* ldsDst, const float2* gsrc, int lane) {
#if __has_builtin(__builtin_amdgcn_tensor_load_to_lds)
    unsigned int ldsOff = (unsigned int)(unsigned long long)(uintptr_t)ldsDst; // LDS aperture: low 32b = offset
    unsigned long long ga = (unsigned long long)(uintptr_t)gsrc;
    ui32x4 g0 = { 1u,                                  // count=1, is_restore=0, gather=0
                  ldsOff,                              // lds_addr [63:32]
                  (unsigned int)(ga & 0xFFFFFFFFull),  // global_addr [95:64]
                  (unsigned int)((ga >> 32) & 0x1FFFFFFull) | (2u << 30) };   // ga[56:32], type=2
    i32x8 g1 = { (int)(3u << 16),                      // data_size=3 (8B), mask=0, no pad/iter
                 (int)((1024u & 0xFFFFu) << 16),       // tensor_dim0[15:0] @ bits 63:48
                 (int)(1u << 16),                      // tensor_dim0[31:16]=0, tensor_dim1=1
                 (int)(1024u << 16),                   // tensor_dim1[31:16]=0, tile_dim0=1024
                 1,                                    // tile_dim1=1, tile_dim2=0
                 1024,                                 // tensor_dim0_stride[31:0]
                 (int)(1024u << 16),                   // stride0[47:32]=0, tensor_dim1_stride[15:0]
                 0 };                                  // tensor_dim1_stride[47:16]=0
    i32x4 gz = { 0, 0, 0, 0 };
#if defined(__clang_major__) && (__clang_major__ >= 23)
    i32x8 gz8 = { 0, 0, 0, 0, 0, 0, 0, 0 };
    __builtin_amdgcn_tensor_load_to_lds(g0, g1, gz, gz, gz8, 0);
#else
    __builtin_amdgcn_tensor_load_to_lds(g0, g1, gz, gz, 0);
#endif
#if __has_builtin(__builtin_amdgcn_s_wait_tensorcnt)
    __builtin_amdgcn_s_wait_tensorcnt(0);
#endif
#else
    for (int i = lane; i < NN; i += 32) ldsDst[i] = gsrc[i];
#endif
}

// ---------------------------------------------------------------------------
// Kernel 1: Hilbert transform per batch.  One wave32 block per batch.
// ---------------------------------------------------------------------------
__device__ void fft1024_stockham(float2* sX, float2* sY, const float2* sTw, int lane, int inverse) {
    float2* src = sX;
    float2* dst = sY;
    int nn = NN, sl = 0, tf = 1;
    for (int st = 0; st < 10; ++st) {             // 2^10 = 1024; even #stages -> result in sX
        int m = nn >> 1;
        int ssm1 = (1 << sl) - 1;
        for (int it = lane; it < (NN >> 1); it += 32) {
            int p = it >> sl;
            int q = it & ssm1;
            int ss = ssm1 + 1;
            float2 a  = src[q + ss * p];
            float2 bb = src[q + ss * (p + m)];
            float2 w  = sTw[p * tf];              // e^{-2pi i p/nn}
            if (inverse) w.y = -w.y;              // conj for inverse
            dst[q + ss * (2 * p)]     = caddf(a, bb);
            dst[q + ss * (2 * p + 1)] = cmulf(csubf(a, bb), w);
        }
        __syncthreads();
        float2* t = src; src = dst; dst = t;
        nn >>= 1; sl++; tf <<= 1;
    }
}

__global__ __launch_bounds__(32)
void hilbert_kernel(const float* __restrict__ pred, float2* __restrict__ pa) {
    const int b = blockIdx.x;
    const int lane = threadIdx.x;
    __shared__ float2 sX[NN];
    __shared__ float2 sY[NN];
    __shared__ float2 sTw[NN];
    for (int i = lane; i < NN; i += 32) {
        float s, c;
        sincosf(-(PI2F / (float)NN) * (float)i, &s, &c);
        sTw[i] = make_float2(c, s);
        sX[i] = make_float2(pred[b * NN + i], 0.0f);
    }
    __syncthreads();
    fft1024_stockham(sX, sY, sTw, lane, 0);       // forward; result in sX
    for (int i = lane; i < NN; i += 32) {         // Hilbert mask: 0 / 1@N/2 / 2 above
        float m = (i > NN / 2) ? 2.0f : ((i == NN / 2) ? 1.0f : 0.0f);
        sX[i].x *= m;
        sX[i].y *= m;
    }
    __syncthreads();
    fft1024_stockham(sX, sY, sTw, lane, 1);       // inverse (unscaled); result in sX
    const float inv = 1.0f / (float)NN;
    for (int i = lane; i < NN; i += 32)
        pa[b * NN + i] = make_float2(sX[i].x * inv, sX[i].y * inv);
}

// ---------------------------------------------------------------------------
// Kernel 2: SHG trace rows + FROG accumulators.  One wave32 block per
// (batch, 8 rows).  Four-step FFT-1024: DFT-16 via WMMA, twiddle,
// DFT-64 Stockham; stream |.|^2 against t_ref.
// ---------------------------------------------------------------------------
__global__ __launch_bounds__(32)
void shg_frog_kernel(const float2* __restrict__ pa, const float* __restrict__ tref,
                     float* __restrict__ acc) {
    const int b = blockIdx.y;
    const int lane = threadIdx.x;
    __shared__ float2 sA[NN];
    __shared__ float2 sG[NN];
    __shared__ float2 sT[NN];
    __shared__ float2 sTw[NN];

    tdm_load_row_f2(sA, pa + (size_t)b * NN, lane);   // DMA; waited on (tensorcnt 0) inside helper before setup below

    for (int i = lane; i < NN; i += 32) {
        float s, c;
        sincosf(-(PI2F / (float)NN) * (float)i, &s, &c);
        sTw[i] = make_float2(c, s);
    }
    // Constant DFT-16 WMMA A-operands (fp16, 16x32 A layout):
    //  Tr = [cos | sin] * [Gr ; Gi],   Ti = [-sin | cos] * [Gr ; Gi]
    const int Mrow = lane & 15;
    const int half = lane >> 4;
    v16h A1{}, A2{};
    for (int idx = 0; idx < 16; ++idx) {
        int v = idx >> 1, j = idx & 1;
        int K = half * 16 + idx;   // contiguous k order (matches B)
        float th = (PI2F / 16.0f) * (float)(Mrow * (K & 15));
        float s, c;
        sincosf(th, &s, &c);
        A1[idx] = (_Float16)((K < 16) ? c : s);
        A2[idx] = (_Float16)((K < 16) ? -s : c);
    }
    __syncthreads();

    float v_max = 0.0f, v_sum = 0.0f, v_dot = 0.0f, v_st = 0.0f, v_s2 = 0.0f;
    const float DT2 = 1.5e-15f * 1.5e-15f;

    // Prefetch first t_ref row (one 128B line per lane covers the 4KB row)
    {
        const float* t0 = tref + ((size_t)b * NN + (size_t)blockIdx.x * ROWS_PER_BLOCK) * NN;
        __builtin_prefetch(t0 + lane * 32, 0, 1);
    }

    for (int r = 0; r < ROWS_PER_BLOCK; ++r) {
        const int d = blockIdx.x * ROWS_PER_BLOCK + r;
        const int delay = d - NN / 2;
        float ps, pc;
        sincosf(3.525f * (float)delay, &ps, &pc);       // 2*W_CENTER*DELTA_TAU = 3.525
        const float2 phase = make_float2(pc, -ps);      // e^{-i*theta}
        for (int i = lane; i < NN; i += 32) {           // g[n] = fftshifted row
            int t = (i + NN / 2) & (NN - 1);
            int tb = (t - delay + 2 * NN) & (NN - 1);
            sG[i] = cmulf(cmulf(sA[t], sA[tb]), phase);
        }
        if (r + 1 < ROWS_PER_BLOCK) {                   // prefetch next t_ref row
            const float* nrow = tref + ((size_t)b * NN + d + 1) * NN;
            __builtin_prefetch(nrow + lane * 32, 0, 1);
        }
        __syncthreads();

        // Step 1: T[k1][n2] = sum_{n1} W16[k1][n1] * G[n1][n2]  (4 column tiles)
        const int col = lane & 15;
        const bool hi = (lane >= 16);
        const float* gcomp = ((const float*)sG) + (hi ? 1 : 0);   // component by address, no cndmask
        for (int tc = 0; tc < 4; ++tc) {
            const int cbase = tc * 16 + col;
            v16h Bm{};                                   // 32x16 B: rows 0-15=Gr, 16-31=Gi
            for (int idx = 0; idx < 16; ++idx)
                Bm[idx] = (_Float16)gcomp[(idx * 64 + cbase) * 2];
            v8f cr = {};
            v8f ci = {};
            cr = __builtin_amdgcn_wmma_f32_16x16x32_f16(false, A1, false, Bm, (short)0, cr, false, false);
            ci = __builtin_amdgcn_wmma_f32_16x16x32_f16(false, A2, false, Bm, (short)0, ci, false, false);
            const int mo = hi ? 8 : 0;                   // C layout: VGPR v -> row v (+8 for hi half)
            for (int v = 0; v < 8; ++v)
                sT[(v + mo) * 64 + cbase] = make_float2(cr[v], ci[v]);
        }
        __syncthreads();

        // Step 2: twiddle  T'[k1][n2] *= e^{-2pi i k1 n2 / 1024}
        for (int i = lane; i < NN; i += 32) {
            int k1 = i >> 6, n2 = i & 63;
            sT[i] = cmulf(sT[i], sTw[(k1 * n2) & (NN - 1)]);
        }
        __syncthreads();

        // Step 3: 16 independent 64-point FFTs (6 radix-2 Stockham stages, sT <-> sG)
        {
            float2* src = sT;
            float2* dst = sG;
            int nn = 64, sl = 0, tf = 16;                // tw index scale = 1024/nn
            for (int st = 0; st < 6; ++st) {
                int m = nn >> 1;
                int ssm1 = (1 << sl) - 1;
                for (int it = lane; it < 512; it += 32) {
                    int row = it >> 5;
                    int op = it & 31;
                    int p = op >> sl;
                    int q = op & ssm1;
                    int ss = ssm1 + 1;
                    int base = row << 6;
                    float2 a  = src[base + q + ss * p];
                    float2 bb = src[base + q + ss * (p + m)];
                    float2 w  = sTw[p * tf];
                    dst[base + q + ss * (2 * p)]     = caddf(a, bb);
                    dst[base + q + ss * (2 * p + 1)] = cmulf(csubf(a, bb), w);
                }
                __syncthreads();
                float2* t = src; src = dst; dst = t;
                nn >>= 1; sl++; tf <<= 1;
            }
        }
        // Result (natural order) in sT:  X[k1 + 16*k2] = sT[k1*64 + k2]

        // Accumulate against t_ref row (coalesced over output index k)
        const float* trow = tref + ((size_t)b * NN + d) * NN;
        for (int k = lane; k < NN; k += 32) {
            int jj = (k + NN / 2) & (NN - 1);            // output fftshift
            float2 Gv = sT[(jj & 15) * 64 + (jj >> 4)];
            float val = (Gv.x * Gv.x + Gv.y * Gv.y) * DT2;
            float tv = trow[k];
            v_sum += val;
            v_dot += val * tv;
            v_st  += tv;
            v_s2  += tv * tv;
            v_max = fmaxf(v_max, val);
        }
    }

    for (int off = 16; off > 0; off >>= 1) {
        v_sum += __shfl_down(v_sum, off, 32);
        v_dot += __shfl_down(v_dot, off, 32);
        v_st  += __shfl_down(v_st,  off, 32);
        v_s2  += __shfl_down(v_s2,  off, 32);
        v_max = fmaxf(v_max, __shfl_down(v_max, off, 32));
    }
    if (lane == 0) {
        float* ab = acc + b * 8;
        atomicMax((unsigned int*)&ab[0], __float_as_uint(v_max));   // val >= 0 -> bit-order ok
        atomicAdd(&ab[1], v_sum);
        atomicAdd(&ab[2], v_dot);
        atomicAdd(&ab[3], v_st);
        atomicAdd(&ab[4], v_s2);
    }
}

// ---------------------------------------------------------------------------
// Kernel 3: masked MSE terms per batch (real/imag/intensity/phase).
// ---------------------------------------------------------------------------
__global__ __launch_bounds__(256)
void mse_kernel(const float2* __restrict__ pa, const float* __restrict__ label,
                float* __restrict__ msebuf) {
    const int b = blockIdx.x;
    const int t = threadIdx.x;
    __shared__ int s_fr, s_lr, s_fi, s_li;
    __shared__ float red[256];
    if (t == 0) { s_fr = NN; s_lr = -1; s_fi = NN; s_li = -1; }
    __syncthreads();
    for (int i = t; i < NN; i += 256) {
        float lr = label[(size_t)b * 2 * NN + i];
        float li = label[(size_t)b * 2 * NN + NN + i];
        if (fabsf(lr) > 0.01f) { atomicMin(&s_fr, i); atomicMax(&s_lr, i); }
        if (fabsf(li) > 0.01f) { atomicMin(&s_fi, i); atomicMax(&s_li, i); }
    }
    __syncthreads();
    const int fr  = (s_lr < 0) ? 0 : s_fr;
    const int lrx = (s_lr < 0) ? (NN - 1) : s_lr;
    const int fi  = (s_li < 0) ? 0 : s_fi;
    const int lix = (s_li < 0) ? (NN - 1) : s_li;
    const int first = min(fr, fi);
    const int last  = max(lrx, lix);
    float tot = 0.0f;
    for (int i = t; i < NN; i += 256) {
        float lr = label[(size_t)b * 2 * NN + i];
        float li = label[(size_t)b * 2 * NN + NN + i];
        float2 p = pa[b * NN + i];
        bool in = (i >= first) && (i < last);
        float pm  = in ? 10.0f : 1.0f;                  // PENALTY
        float phm = in ? 1.0f : 0.0f;
        float dr = p.x - lr, di = p.y - li;
        float dI = (p.x * p.x + p.y * p.y) - (lr * lr + li * li);
        float dph = atan2f(p.y, p.x) - atan2f(li, lr);
        tot += (dr * dr + di * di + dI * dI) * pm + dph * dph * phm;
    }
    red[t] = tot;
    __syncthreads();
    for (int s = 128; s > 0; s >>= 1) {
        if (t < s) red[t] += red[t + s];
        __syncthreads();
    }
    if (t == 0) msebuf[b] = red[0] / ((float)NN * 4.0f);   // /N mean, /MSE_WSUM
}

// ---------------------------------------------------------------------------
// Kernel 4: final scalar = mean_b( mse_b + frog_b ).
// frog = |Sraw/M - mu*St| / N  with mu = (Sdot/M)/S2  (norm = N^2 since
// normalized max == 1), matching jnp.where(r==0, 0, sqrt(r/norm)).
// ---------------------------------------------------------------------------
__global__ __launch_bounds__(32)
void final_kernel(const float* __restrict__ acc, const float* __restrict__ msebuf,
                  float* __restrict__ out) {
    const int b = threadIdx.x;
    const float* ab = acc + b * 8;
    float M    = __uint_as_float(((const unsigned int*)ab)[0]);
    float Sraw = ab[1], Sdot = ab[2], St = ab[3], S2 = ab[4];
    float mu = (Sdot / M) / S2;
    float diff = Sraw / M - mu * St;
    float r = diff * diff;
    float frog = (r == 0.0f) ? 0.0f : (sqrtf(r) / (float)NN);
    float loss = msebuf[b] + frog;
    for (int off = 16; off > 0; off >>= 1) loss += __shfl_down(loss, off, 32);
    if (b == 0) out[0] = loss / (float)BB;
}

// ---------------------------------------------------------------------------
extern "C" void kernel_launch(void* const* d_in, const int* in_sizes, int n_in,
                              void* d_out, int out_size, void* d_ws, size_t ws_size,
                              hipStream_t stream) {
    const float* pred  = (const float*)d_in[0];   // (B, N)
    const float* label = (const float*)d_in[1];   // (B, 2N)
    const float* tref  = (const float*)d_in[2];   // (B, 1, N, N)
    float* out = (float*)d_out;

    float2* pa    = (float2*)d_ws;                                   // B*N complex
    float*  acc   = (float*)((char*)d_ws + (size_t)BB * NN * sizeof(float2));  // B*8
    float*  mseb  = acc + BB * 8;                                    // B

    hipMemsetAsync(acc, 0, (size_t)(BB * 8 + BB) * sizeof(float), stream);

    hilbert_kernel<<<BB, 32, 0, stream>>>(pred, pa);
    shg_frog_kernel<<<dim3(NN / ROWS_PER_BLOCK, BB), 32, 0, stream>>>(pa, tref, acc);
    mse_kernel<<<BB, 256, 0, stream>>>(pa, label, mseb);
    final_kernel<<<1, 32, 0, stream>>>(acc, mseb, out);
}
